// Lstm_38388417692031
// MI455X (gfx1250) — matrix operand order repaired and checked by reading, not compiled, not run on hardware
//
#include <hip/hip_runtime.h>
#include <hip/hip_bf16.h>

// ---------------------------------------------------------------------------
// LSTM decoder for MI455X (gfx1250, wave32, WMMA, TDM).
//
//  1) lstm_init_kernel   : publish h0 (f32+bf16), zero grid barrier.
//  2) gather_x_kernel    : X_bf16[4096,512]  = bf16(emb[src[t],:])   (4MB)
//     cvt_f32_bf16_kernel: W_ih -> bf16 (4MB), W_hh -> bf16 (8MB)
//  3) lstm_xproj_kernel  : XP = X_bf16 @ W_ih_bf16^T, pure-bf16 WMMA GEMM,
//                          2x2 wave blocking (32x32 out per wave), f32 accum.
//  4) lstm_recur_kernel  : 32 persistent WGs; W_hh slice (260KB, padded rows)
//                          pulled into LDS by the Tensor Data Mover
//                          (tensor_load_to_lds + s_wait_tensorcnt), then the
//                          whole sequential scan runs out of LDS with a
//                          generation-counter grid barrier per step.
// ---------------------------------------------------------------------------

typedef __bf16 bf16_t;
typedef bf16_t bf16x8  __attribute__((ext_vector_type(8)));
typedef bf16_t bf16x16 __attribute__((ext_vector_type(16)));
typedef float  f32x4   __attribute__((ext_vector_type(4)));
typedef float  f32x8   __attribute__((ext_vector_type(8)));
typedef unsigned int u32x4 __attribute__((ext_vector_type(4)));
typedef int    i32x4   __attribute__((ext_vector_type(4)));
typedef int    i32x8   __attribute__((ext_vector_type(8)));

#define TQ   4096   // B*L timestep rows
#define NQ   4096   // 4*H gate columns
#define KQ   512    // E
#define HDIM 1024   // H
#define ODIM 16     // O
#define NWG  32     // persistent workgroups in recurrent kernel
#define HC   (HDIM / NWG)   // 32 h elements per WG
// LDS W row: 1024 bf16 = 2048B, TDM pads +16B after every 1024B stored
// (pad_interval=7 -> 256 DWORDs, pad_amount=3 -> 4 DWORDs). Row stride 2080B.
#define WROWB 2080u

#if defined(__has_builtin)
#if __has_builtin(__builtin_amdgcn_tensor_load_to_lds)
#define HAVE_TDM 1
#endif
#endif

__device__ __forceinline__ bf16x16 cat16(bf16x8 lo, bf16x8 hi) {
  return __builtin_shufflevector(lo, hi, 0, 1, 2, 3, 4, 5, 6, 7,
                                         8, 9, 10, 11, 12, 13, 14, 15);
}
__device__ __forceinline__ float sigf(float x) { return 1.0f / (1.0f + __expf(-x)); }

// Byte offset of (row r, col c) in the padded LDS W slice (pad after col 512).
__device__ __forceinline__ unsigned wrow_off(int r, int c) {
  return (unsigned)r * WROWB + (unsigned)c * 2u + (unsigned)((c >> 9) << 4);
}

// ---------------------------------------------------------------------------
// Kernel 1: init h buffers + grid barrier counter.
// ---------------------------------------------------------------------------
__global__ void lstm_init_kernel(const float* __restrict__ hidden,
                                 float* __restrict__ h_f32,
                                 bf16_t* __restrict__ h_b16,
                                 unsigned* __restrict__ bar) {
  int i = blockIdx.x * blockDim.x + threadIdx.x;
  if (i < HDIM) {
    float v = hidden[i];
    h_f32[i] = v;
    h_b16[i] = (bf16_t)v;
  }
  if (i == 0) *bar = 0u;
}

// ---------------------------------------------------------------------------
// Prep: embedding gather -> bf16 rows; generic f32 -> bf16 convert.
// ---------------------------------------------------------------------------
__global__ __launch_bounds__(128) void gather_x_kernel(
    const int* __restrict__ src, const float* __restrict__ emb,
    bf16_t* __restrict__ Xb) {
  const int row = blockIdx.x;
  const float* e = emb + (size_t)src[row] * KQ;
  bf16_t* d = Xb + (size_t)row * KQ;
  const int c = threadIdx.x * 4;
  f32x4 v = *(const f32x4*)(e + c);
  d[c + 0] = (bf16_t)v[0]; d[c + 1] = (bf16_t)v[1];
  d[c + 2] = (bf16_t)v[2]; d[c + 3] = (bf16_t)v[3];
}

__global__ __launch_bounds__(256) void cvt_f32_bf16_kernel(
    const float* __restrict__ s, bf16_t* __restrict__ d) {
  const size_t i = ((size_t)blockIdx.x * blockDim.x + threadIdx.x) * 4;
  f32x4 v = *(const f32x4*)(s + i);
  d[i + 0] = (bf16_t)v[0]; d[i + 1] = (bf16_t)v[1];
  d[i + 2] = (bf16_t)v[2]; d[i + 3] = (bf16_t)v[3];
}

// ---------------------------------------------------------------------------
// Kernel 2: XP = X @ W_ih^T, pure bf16 WMMA, 2x2 wave blocking.
// A layout (16-bit 16x32, contiguous k like B): lane m<16 holds row m elems {K0..15};
// lane m+16 holds row m elems {K16..31}.
// B layout (32x16): lane l holds column N=l&15, K=(l>>4)*16+j contiguous.
// ---------------------------------------------------------------------------
__global__ __launch_bounds__(256) void lstm_xproj_kernel(
    const bf16_t* __restrict__ Xb, const bf16_t* __restrict__ Wb,
    float* __restrict__ XP) {
  const int wave = threadIdx.x >> 5;
  const int lane = threadIdx.x & 31;
  const int lm   = lane & 15;
  const bool hi  = lane >= 16;

  const int wt = blockIdx.x * 8 + wave;   // 128x128 wave-tiles of 32x32
  const int tm = (wt >> 7) << 5;          // timestep base
  const int tn = (wt & 127) << 5;         // gate-column base

  const bf16_t* a0 = Xb + (size_t)(tm + lm) * KQ;
  const bf16_t* a1 = Xb + (size_t)(tm + 16 + lm) * KQ;
  const bf16_t* b0 = Wb + (size_t)(tn + lm) * KQ + (hi ? 16 : 0);
  const bf16_t* b1 = Wb + (size_t)(tn + 16 + lm) * KQ + (hi ? 16 : 0);

  f32x8 acc00 = {}, acc01 = {}, acc10 = {}, acc11 = {};
  const int o0 = hi ? 16 : 0, o1 = hi ? 24 : 8;
#pragma unroll 4
  for (int kc = 0; kc < KQ; kc += 32) {
    bf16x16 A0 = cat16(*(const bf16x8*)(a0 + kc + o0), *(const bf16x8*)(a0 + kc + o1));
    bf16x16 A1 = cat16(*(const bf16x8*)(a1 + kc + o0), *(const bf16x8*)(a1 + kc + o1));
    bf16x16 B0 = cat16(*(const bf16x8*)(b0 + kc),      *(const bf16x8*)(b0 + kc + 8));
    bf16x16 B1 = cat16(*(const bf16x8*)(b1 + kc),      *(const bf16x8*)(b1 + kc + 8));
    acc00 = __builtin_amdgcn_wmma_f32_16x16x32_bf16(false, A0, false, B0, (short)0, acc00, false, false);
    acc01 = __builtin_amdgcn_wmma_f32_16x16x32_bf16(false, A0, false, B1, (short)0, acc01, false, false);
    acc10 = __builtin_amdgcn_wmma_f32_16x16x32_bf16(false, A1, false, B0, (short)0, acc10, false, false);
    acc11 = __builtin_amdgcn_wmma_f32_16x16x32_bf16(false, A1, false, B1, (short)0, acc11, false, false);
  }

  // D layout: VGPR r, lanes 0-15 -> (M=r, N=lane); lanes 16-31 -> (M=8+r).
  const int mrow = tm + (hi ? 8 : 0);
  float* p00 = XP + (size_t)mrow * NQ + tn + lm;
#pragma unroll
  for (int r = 0; r < 8; ++r) {
    p00[(size_t)r * NQ]             = acc00[r];
    p00[(size_t)r * NQ + 16]        = acc01[r];
    p00[(size_t)(r + 16) * NQ]      = acc10[r];
    p00[(size_t)(r + 16) * NQ + 16] = acc11[r];
  }
}

// ---------------------------------------------------------------------------
// TDM: issue a 2D tile load global->LDS (bf16 elements) with row padding.
// D# packing per CDNA5 ISA §8.3/8.4 (group0: count/lds/global/type,
// group1: data_size|pad | dims | strides). Semantics are not hardware
// verified (compile-only environment); fallback path covers non-TDM builds.
// ---------------------------------------------------------------------------
#if defined(HAVE_TDM)
__device__ __forceinline__ void tdm_load_2d_bf16(
    unsigned lds_byte, const void* gptr, unsigned rows, unsigned cols,
    unsigned row_stride_elems) {
  const unsigned long long ga = (unsigned long long)(uintptr_t)gptr;
  u32x4 g0;
  g0[0] = 1u;                                        // count=1, user D#
  g0[1] = lds_byte;                                  // lds_addr
  g0[2] = (unsigned)(ga & 0xffffffffu);              // global_addr[31:0]
  g0[3] = (unsigned)((ga >> 32) & 0x1ffffffu) | (2u << 30);  // addr[56:32]|type=2
  i32x8 g1;
  g1[0] = (int)((1u << 16)        // data_size = 1 -> 2 bytes
              | (1u << 20)        // pad_enable
              | (7u << 22)        // pad_interval: 256 DWORDs (1024B)
              | (3u << 25));      // pad_amount:   4 DWORDs  (16B)
  g1[1] = (int)((cols & 0xffffu) << 16);                       // tensor_dim0 lo
  g1[2] = (int)(((cols >> 16) & 0xffffu) | ((rows & 0xffffu) << 16)); // d0 hi|d1 lo
  g1[3] = (int)(((rows >> 16) & 0xffffu) | ((cols & 0xffffu) << 16)); // d1 hi|tile_d0
  g1[4] = (int)(rows & 0xffffu);                               // tile_d1 (tile_d2=0)
  g1[5] = (int)row_stride_elems;                               // d0_stride lo32
  g1[6] = 0;                                                   // d0_stride hi|d1_stride lo
  g1[7] = 0;
  const i32x4 z4 = {0, 0, 0, 0};
#if defined(__clang_major__) && __clang_major__ >= 23
  const i32x8 z8 = {0, 0, 0, 0, 0, 0, 0, 0};
  __builtin_amdgcn_tensor_load_to_lds(g0, g1, z4, z4, z8, 0);
#else
  __builtin_amdgcn_tensor_load_to_lds(g0, g1, z4, z4, 0);
#endif
}
#endif

// ---------------------------------------------------------------------------
// Kernel 3: persistent recurrent scan. 32 WGs x 256 threads (8 waves).
// WG k owns h[k*32 .. k*32+32); gate rows {q*1024 + k*32 + r}, q in {i,f,g,o}.
// W_hh slice: 128 rows x 1024 bf16, padded stride 2080B, in LDS for the
// whole kernel (~260KB of the WGP's 320KB). Loaded by TDM (4 band tiles).
// ---------------------------------------------------------------------------
__global__ __launch_bounds__(256) void lstm_recur_kernel(
    const int* __restrict__ nsteps, const float* __restrict__ cell0,
    const bf16_t* __restrict__ Whhb, const float* __restrict__ XP,
    const float* __restrict__ fcw, const float* __restrict__ fcb,
    float* __restrict__ h_f32, bf16_t* __restrict__ h_b16,
    unsigned* __restrict__ bar, float* __restrict__ out) {
  extern __shared__ char smem[];
  char*   sW   = smem;                                  // 128 * 2080 bytes
  bf16_t* sh   = (bf16_t*)(smem + 128 * WROWB);         // 1024 bf16
  float*  gbuf = (float*)(smem + 128 * WROWB + 2048);   // 128 f32
  float*  cs   = gbuf + 128;                            // 32 f32

  const int tid = threadIdx.x;
  const int wg  = blockIdx.x;
  const int wave = tid >> 5, lane = tid & 31, lm = lane & 15;
  const bool hiL = lane >= 16;

  // ---- Preload W_hh slice into LDS via the Tensor Data Mover.
#if defined(HAVE_TDM)
  if (wave == 0) {
#pragma unroll
    for (int q = 0; q < 4; ++q) {
      tdm_load_2d_bf16((unsigned)(q * 32) * WROWB,
                       Whhb + (size_t)(q * HDIM + wg * HC) * HDIM,
                       /*rows=*/HC, /*cols=*/HDIM, /*stride=*/HDIM);
    }
    __builtin_amdgcn_s_wait_tensorcnt(0);
  }
#else
  {
    const int r     = tid >> 1;            // local row 0..127
    const int halfc = (tid & 1) * 512;     // pad-group aligned column half
    const int q     = r >> 5, rr = r & 31;
    const bf16_t* g = Whhb + (size_t)(q * HDIM + wg * HC + rr) * HDIM + halfc;
    bf16_t* d = (bf16_t*)(sW + wrow_off(r, halfc));
    for (int c = 0; c < 512; c += 8) *(bf16x8*)(d + c) = *(const bf16x8*)(g + c);
  }
#endif
  if (tid < HC) cs[tid] = cell0[wg * HC + tid];
  __syncthreads();

  const int q    = wave >> 1;            // gate band for this wave
  const int half = wave & 1;             // which 16 of the 32 band rows
  const int lr0  = q * 32 + half * 16;   // local gate-row base (== gbuf base)
  const char* wbase = sW + (size_t)(lr0 + lm) * WROWB;
  const int nglob   = q * HDIM + wg * HC + half * 16 + lm;
  const int o0 = hiL ? 16 : 0, o1 = hiL ? 24 : 8, bcol0 = hiL ? 16 : 0;

  int NS[8];
#pragma unroll
  for (int b = 0; b < 8; ++b) NS[b] = nsteps[b];

  unsigned gen = 0;
  for (int b = 0; b < 8; ++b) {
    for (int t = 0; t < NS[b]; ++t) {
      const size_t row = (size_t)(b * 512 + t);
      const float* xpp = XP + row * NQ + nglob;
      __builtin_prefetch(xpp, 0, 0);   // global_prefetch: warm XP addend

      // Stage current h (bf16) into LDS: 128 threads x 16B.
      if (tid < 128) ((bf16x8*)sh)[tid] = ((const bf16x8*)h_b16)[tid];
      __syncthreads();

      // gates[lr0+lm] = h . W_hh_row via 32 chained WMMAs (A = h broadcast
      // into all 16 M rows -> every D row equals the wanted dot product).
      f32x8 acc = {};
#pragma unroll 4
      for (int kc = 0; kc < HDIM; kc += 32) {
        const bf16_t* ha = sh + kc;
        const int col = kc + bcol0;
        const bf16_t* wb = (const bf16_t*)(wbase + col * 2 + ((col >> 9) << 4));
        acc = __builtin_amdgcn_wmma_f32_16x16x32_bf16(
            false, cat16(*(const bf16x8*)(ha + o0), *(const bf16x8*)(ha + o1)),
            false, cat16(*(const bf16x8*)wb,        *(const bf16x8*)(wb + 8)),
            (short)0, acc, false, false);
      }
      if (lane < 16) gbuf[lr0 + lm] = acc[0] + *xpp;
      __syncthreads();

      // Gate math + publish h slice.
      if (tid < HC) {
        float ig = gbuf[tid], fg = gbuf[32 + tid];
        float gg = gbuf[64 + tid], og = gbuf[96 + tid];
        float cn = sigf(fg) * cs[tid] + sigf(ig) * tanhf(gg);
        float hn = sigf(og) * tanhf(cn);
        cs[tid] = cn;
        h_f32[wg * HC + tid] = hn;
        h_b16[wg * HC + tid] = (bf16_t)hn;
        __threadfence();
      }
      __syncthreads();

      // Grid barrier: generation counter in L2. All 32 WGs are resident
      // (one per WGP due to the ~260KB LDS footprint), so spinning is safe.
      ++gen;
      if (tid == 0) {
        __hip_atomic_fetch_add(bar, 1u, __ATOMIC_ACQ_REL, __HIP_MEMORY_SCOPE_AGENT);
        while (__hip_atomic_load(bar, __ATOMIC_ACQUIRE, __HIP_MEMORY_SCOPE_AGENT)
               < gen * (unsigned)NWG)
          __builtin_amdgcn_s_sleep(2);
      }
      __syncthreads();
    }
  }

  // ---- Outputs: d_out = [dense(16) | h(1024) | c(1024)].
  if (tid < HC) {
    out[ODIM + wg * HC + tid]        = h_f32[wg * HC + tid];
    out[ODIM + HDIM + wg * HC + tid] = cs[tid];
  }
  if (wg == 0 && tid < ODIM) {
    float s = fcb[tid];
    const float* w = fcw + (size_t)tid * HDIM;
    for (int k = 0; k < HDIM; ++k) s += h_f32[k] * w[k];
    out[tid] = s;
  }
}

// ---------------------------------------------------------------------------
extern "C" void kernel_launch(void* const* d_in, const int* in_sizes, int n_in,
                              void* d_out, int out_size, void* d_ws, size_t ws_size,
                              hipStream_t stream) {
  const int*   src    = (const int*)d_in[0];
  const int*   nsteps = (const int*)d_in[1];
  const float* hidden = (const float*)d_in[2];
  const float* cell   = (const float*)d_in[3];
  const float* emb    = (const float*)d_in[4];
  const float* Wih    = (const float*)d_in[5];
  const float* Whh    = (const float*)d_in[6];
  const float* fcw    = (const float*)d_in[7];
  const float* fcb    = (const float*)d_in[8];
  float* out = (float*)d_out;

  char* ws = (char*)d_ws;
  size_t off = 0;
  float*  XP   = (float*)(ws + off);  off += (size_t)TQ * NQ * sizeof(float);   // 64MB
  bf16_t* Xb   = (bf16_t*)(ws + off); off += (size_t)TQ * KQ * sizeof(bf16_t);  // 4MB
  bf16_t* Wihb = (bf16_t*)(ws + off); off += (size_t)NQ * KQ * sizeof(bf16_t);  // 4MB
  bf16_t* Whhb = (bf16_t*)(ws + off); off += (size_t)NQ * HDIM * sizeof(bf16_t);// 8MB
  bf16_t* h_b16 = (bf16_t*)(ws + off); off += HDIM * sizeof(bf16_t);
  float*  h_f32 = (float*)(ws + off);  off += HDIM * sizeof(float);
  unsigned* bar = (unsigned*)(ws + off);

  lstm_init_kernel<<<4, 256, 0, stream>>>(hidden, h_f32, h_b16, bar);

  // Operand preparation (one-time f32 -> bf16; gather fused for X).
  gather_x_kernel<<<TQ, 128, 0, stream>>>(src, emb, Xb);
  cvt_f32_bf16_kernel<<<(NQ * KQ / 4) / 256, 256, 0, stream>>>(Wih, Wihb);
  cvt_f32_bf16_kernel<<<(NQ * HDIM / 4) / 256, 256, 0, stream>>>(Whh, Whhb);

  // 16384 32x32 wave-tiles, 8 waves per 256-thread block.
  lstm_xproj_kernel<<<(TQ / 32) * (NQ / 32) / 8, 256, 0, stream>>>(Xb, Wihb, XP);

  const size_t smbytes = (size_t)128 * WROWB          // W_hh slice (padded bf16)
                       + 2048                         // h staging (bf16)
                       + (128 + 32) * sizeof(float);  // gbuf + c slice
  lstm_recur_kernel<<<NWG, 256, smbytes, stream>>>(
      nsteps, cell, Whhb, XP, fcw, fcb, h_f32, h_b16, bar, out);
}
